// MambaBlock_53652731462300
// MI455X (gfx1250) — matrix operand order repaired and checked by reading, not compiled, not run on hardware
//
#include <hip/hip_runtime.h>
#include <hip/hip_bf16.h>
#include <stddef.h>
#include <stdint.h>

// ---------------------------------------------------------------------------
// Mamba block for MI455X (gfx1250, wave32):
//  - WMMA bf16 16x16x32, f32 accumulate for all GEMMs
//  - global_load_async_to_lds_b128 (ASYNCcnt) double-buffered tile staging
//  - register-resident selective scan (one thread per (b,e) channel)
// H=1024, E=2048, N=16, R=64, B=2, L=2048 -> MT = 4096 token rows
// ---------------------------------------------------------------------------

#define H_DIM 1024
#define E_DIM 2048
#define N_ST  16
#define R_DIM 64
#define B_SZ  2
#define L_SEQ 2048
#define MT    (B_SZ * L_SEQ)

typedef __attribute__((ext_vector_type(16))) __bf16 v16bf;
typedef __attribute__((ext_vector_type(8)))  __bf16 v8bf;
typedef __attribute__((ext_vector_type(8)))  float  v8f;

// fp32 -> bf16 round-to-nearest-even
__device__ __forceinline__ __bf16 f2bf(float f) {
  unsigned int u = __builtin_bit_cast(unsigned int, f);
  u = (u + 0x7FFFu + ((u >> 16) & 1u)) >> 16;
  unsigned short s = (unsigned short)u;
  return __builtin_bit_cast(__bf16, s);
}

__device__ __forceinline__ float silu_f(float v) {
  return v * (1.0f / (1.0f + __expf(-v)));
}
__device__ __forceinline__ float softplus_f(float v) {
  return (v > 20.0f) ? v : log1pf(__expf(v));
}

// --- gfx1250 async global->LDS copy (VGLOBAL GLOBAL_LOAD_ASYNC_TO_LDS_B128,
//     tracked by ASYNCcnt; see CDNA5 ISA) ----------------
__device__ __forceinline__ void async_copy_b128(const void* lds_dst,
                                                const void* gsrc) {
  // Generic LDS address: addr[31:0] is the LDS byte offset (ISA aperture map).
  unsigned lds_off = (unsigned)(uintptr_t)lds_dst;
  unsigned long long ga = (unsigned long long)(uintptr_t)gsrc;
  asm volatile("global_load_async_to_lds_b128 %0, %1, off"
               :: "v"(lds_off), "v"(ga)
               : "memory");
}
__device__ __forceinline__ void wait_async0() {
  asm volatile("s_wait_asynccnt 0" ::: "memory");
}

// ---------------------------------------------------------------------------
// LayerNorm over H=1024, one block (256 threads) per token; bf16 output
// ---------------------------------------------------------------------------
__global__ void ln_kernel(const float* __restrict__ x,
                          const float* __restrict__ gamma,
                          const float* __restrict__ beta,
                          __bf16* __restrict__ xn) {
  const int row = blockIdx.x;
  const int tid = threadIdx.x;
  const float* xr = x + (size_t)row * H_DIM;

  __shared__ float ssum[256];
  __shared__ float ssq[256];

  float s = 0.f, q = 0.f;
#pragma unroll
  for (int i = 0; i < H_DIM / 256; ++i) {
    float v = xr[tid + i * 256];
    s += v; q += v * v;
  }
  ssum[tid] = s; ssq[tid] = q;
  __syncthreads();
  for (int st = 128; st > 0; st >>= 1) {
    if (tid < st) { ssum[tid] += ssum[tid + st]; ssq[tid] += ssq[tid + st]; }
    __syncthreads();
  }
  const float mean = ssum[0] * (1.0f / H_DIM);
  const float var  = ssq[0] * (1.0f / H_DIM) - mean * mean;
  const float rstd = rsqrtf(var + 1e-5f);

  __bf16* xo = xn + (size_t)row * H_DIM;
#pragma unroll
  for (int i = 0; i < H_DIM / 256; ++i) {
    int c = tid + i * 256;
    xo[c] = f2bf((xr[c] - mean) * rstd * gamma[c] + beta[c]);
  }
}

// ---------------------------------------------------------------------------
// Weight prep: Wt[n*K + k] = bf16(W[k*N + n]), rows n < Npad zero-padded.
// 32x32 LDS tile transpose, coalesced both directions. grid=(K/32, Npad/32).
// ---------------------------------------------------------------------------
__global__ void cast_transpose(const float* __restrict__ W,
                               __bf16* __restrict__ Wt,
                               int K, int N) {
  __shared__ float t[32][33];
  const int k0 = blockIdx.x * 32;
  const int n0 = blockIdx.y * 32;
  const int tid = threadIdx.x;
#pragma unroll
  for (int i = 0; i < 4; ++i) {
    int idx = tid + i * 256;
    int r = idx >> 5;            // k local
    int c = idx & 31;            // n local
    int n = n0 + c;
    t[r][c] = (n < N) ? W[(size_t)(k0 + r) * N + n] : 0.0f;
  }
  __syncthreads();
#pragma unroll
  for (int i = 0; i < 4; ++i) {
    int idx = tid + i * 256;
    int nl = idx >> 5;
    int kl = idx & 31;
    Wt[(size_t)(n0 + nl) * K + (k0 + kl)] = f2bf(t[kl][nl]);
  }
}

// ---------------------------------------------------------------------------
// WMMA GEMM: out = A[M,K](bf16) @ W[K,N] + bias, W given pre-transposed bf16
// Wt[Npad,K]. Block: 256 thr = 8 waves; tile 128x64, BK=32, double-buffered
// LDS filled via async b128 copies. Wave w owns rows [16w,16w+16) x 4 n-tiles.
// MODE 0: out0(f32)=v | 1: split->silu: u(f32+bf16), sg(f32) | 2: softplus
// MODE 3: v + resid | 4: outb(bf16)=v
// Requires M%128==0, K%32==0, Npad%64==0 (epilogue guards col<N).
// ---------------------------------------------------------------------------
#define BM 128
#define BN 64
#define BK 32

template <int MODE>
__global__ void wmma_gemm(const __bf16* __restrict__ A,
                          const __bf16* __restrict__ Wt,
                          const float* __restrict__ bias,
                          float* __restrict__ out0,
                          float* __restrict__ out1,
                          __bf16* __restrict__ outb,
                          const float* __restrict__ resid,
                          int M, int N, int K) {
  __shared__ __bf16 As[2][BM][BK];   // 2 x 8 KB
  __shared__ __bf16 Bs[2][BN][BK];   // 2 x 4 KB

  const int tid  = threadIdx.x;
  const int wave = tid >> 5;
  const int lane = tid & 31;
  const int m0   = blockIdx.x * BM;
  const int n0   = blockIdx.y * BN;

  // Per-thread staging coordinates (16B chunks)
  const int ar0 = tid >> 2,          akc0 = (tid & 3) * 8;          // A chunk 0
  const int ar1 = (tid + 256) >> 2,  akc1 = ((tid + 256) & 3) * 8;  // A chunk 1
  const int bn  = tid >> 2,          bkc  = (tid & 3) * 8;          // B chunk

  v8f acc[4];
#pragma unroll
  for (int nt = 0; nt < 4; ++nt)
#pragma unroll
    for (int i = 0; i < 8; ++i) acc[nt][i] = 0.0f;

  // stage(k0) into buffer `buf` with async b128 copies
  auto stage = [&](int buf, int k0) {
    async_copy_b128(&As[buf][ar0][akc0], A  + (size_t)(m0 + ar0) * K + k0 + akc0);
    async_copy_b128(&As[buf][ar1][akc1], A  + (size_t)(m0 + ar1) * K + k0 + akc1);
    async_copy_b128(&Bs[buf][bn ][bkc ], Wt + (size_t)(n0 + bn ) * K + k0 + bkc);
  };

  stage(0, 0);
  wait_async0();
  __syncthreads();

  int buf = 0;
  for (int k0 = 0; k0 < K; k0 += BK) {
    if (k0 + BK < K) stage(buf ^ 1, k0 + BK);   // overlap copy with compute

    // A fragment (ISA 16-bit A 16x32 wave32 layout)
    v16bf afrag;
    {
      int m   = (wave << 4) + (lane & 15);
      int kb0 = (lane < 16) ? 0 : 8;
      v8bf lo = *(const v8bf*)&As[buf][m][kb0];
      v8bf hi = *(const v8bf*)&As[buf][m][kb0 + 16];
#pragma unroll
      for (int i = 0; i < 8; ++i) { afrag[i] = lo[i]; afrag[8 + i] = hi[i]; }
    }
#pragma unroll
    for (int nt = 0; nt < 4; ++nt) {
      int n  = (nt << 4) + (lane & 15);
      int kb = (lane < 16) ? 0 : 8;
      const v8bf* bp = (const v8bf*)&Bs[buf][n][kb];
      v8bf b0 = bp[0], b1 = bp[2];
      v16bf bfrag;
#pragma unroll
      for (int i = 0; i < 8; ++i) { bfrag[i] = b0[i]; bfrag[8 + i] = b1[i]; }
      acc[nt] = __builtin_amdgcn_wmma_f32_16x16x32_bf16(
          false, afrag, false, bfrag, (short)0, acc[nt], false, false);
    }

    wait_async0();      // next tile landed in LDS (this wave's copies)
    __syncthreads();    // all waves' copies landed; reads of `buf` done
    buf ^= 1;
  }

  // Epilogue (C/D layout: VGPR r -> M=r lanes 0-15, M=8+r lanes 16-31)
  const int halfN = N >> 1;
#pragma unroll
  for (int nt = 0; nt < 4; ++nt) {
#pragma unroll
    for (int r = 0; r < 8; ++r) {
      int mrow = m0 + (wave << 4) + ((lane < 16) ? r : (r + 8));
      int col  = n0 + (nt << 4) + (lane & 15);
      if (col >= N) continue;
      float v = acc[nt][r] + bias[col];
      if constexpr (MODE == 0) {
        out0[(size_t)mrow * N + col] = v;
      } else if constexpr (MODE == 1) {
        float sv = silu_f(v);
        if (col < halfN) {
          out0[(size_t)mrow * halfN + col] = sv;        // u  (f32, for scan)
          outb[(size_t)mrow * halfN + col] = f2bf(sv);  // u  (bf16, GEMM A)
        } else {
          out1[(size_t)mrow * halfN + (col - halfN)] = sv;  // silu(gate)
        }
      } else if constexpr (MODE == 2) {
        out0[(size_t)mrow * N + col] = softplus_f(v);
      } else if constexpr (MODE == 3) {
        out0[(size_t)mrow * N + col] = v + resid[(size_t)mrow * N + col];
      } else {  // MODE == 4
        outb[(size_t)mrow * N + col] = f2bf(v);
      }
    }
  }
}

// ---------------------------------------------------------------------------
// Selective scan: one thread per (b,e); h[16], A-row in VGPRs; 2048 serial
// steps; delta/u coalesce over e, B/C rows broadcast through L0/L2.
// Fuses y = (scan + u*D) * silu_gate, emits bf16 for the final WMMA GEMM.
// ---------------------------------------------------------------------------
__global__ void scan_kernel(const float* __restrict__ u,
                            const float* __restrict__ delta,
                            const float* __restrict__ Bm,
                            const float* __restrict__ Cm,
                            const float* __restrict__ A_log,
                            const float* __restrict__ Dvec,
                            const float* __restrict__ sg,
                            __bf16* __restrict__ y) {
  const int g = blockIdx.x * blockDim.x + threadIdx.x;
  if (g >= B_SZ * E_DIM) return;
  const int e = g % E_DIM;
  const int b = g / E_DIM;

  float Arow[N_ST], h[N_ST];
#pragma unroll
  for (int n = 0; n < N_ST; ++n) {
    Arow[n] = -__expf(A_log[(size_t)e * N_ST + n]);
    h[n] = 0.0f;
  }
  const float Dv = Dvec[e];

  const size_t rowBase = (size_t)b * L_SEQ;
  const float4* Bp = (const float4*)Bm + rowBase * (N_ST / 4);
  const float4* Cp = (const float4*)Cm + rowBase * (N_ST / 4);

  for (int t = 0; t < L_SEQ; ++t) {
    const size_t idx = (rowBase + t) * E_DIM + e;
    const float du = delta[idx];
    const float xv = u[idx];

    float bb[N_ST], cc[N_ST];
#pragma unroll
    for (int q = 0; q < N_ST / 4; ++q) {
      float4 b4 = Bp[(size_t)t * (N_ST / 4) + q];
      float4 c4 = Cp[(size_t)t * (N_ST / 4) + q];
      bb[4*q+0] = b4.x; bb[4*q+1] = b4.y; bb[4*q+2] = b4.z; bb[4*q+3] = b4.w;
      cc[4*q+0] = c4.x; cc[4*q+1] = c4.y; cc[4*q+2] = c4.z; cc[4*q+3] = c4.w;
    }

    const float dux = du * xv;
    float accv = 0.0f;
#pragma unroll
    for (int n = 0; n < N_ST; ++n) {
      float ad = __expf(du * Arow[n]);
      h[n] = fmaf(ad, h[n], dux * bb[n]);
      accv = fmaf(cc[n], h[n], accv);
    }
    y[idx] = f2bf((accv + xv * Dv) * sg[idx]);
  }
}

// ---------------------------------------------------------------------------
// Launcher
// ---------------------------------------------------------------------------
extern "C" void kernel_launch(void* const* d_in, const int* in_sizes, int n_in,
                              void* d_out, int out_size, void* d_ws, size_t ws_size,
                              hipStream_t stream) {
  (void)in_sizes; (void)n_in; (void)out_size; (void)ws_size;

  const float* x        = (const float*)d_in[0];
  const float* ln_gamma = (const float*)d_in[1];
  const float* ln_beta  = (const float*)d_in[2];
  const float* W_in     = (const float*)d_in[3];
  const float* b_in     = (const float*)d_in[4];
  const float* W_delta  = (const float*)d_in[5];
  const float* b_delta  = (const float*)d_in[6];
  const float* W_dt     = (const float*)d_in[7];
  const float* b_dt     = (const float*)d_in[8];
  const float* W_B      = (const float*)d_in[9];
  const float* b_B      = (const float*)d_in[10];
  const float* W_C      = (const float*)d_in[11];
  const float* b_C      = (const float*)d_in[12];
  const float* A_log    = (const float*)d_in[13];
  const float* Dvec     = (const float*)d_in[14];
  const float* W_out    = (const float*)d_in[15];
  const float* b_out    = (const float*)d_in[16];
  float* out = (float*)d_out;

  // ---- workspace carve-up -------------------------------------------------
  // fp32 region first, then bf16 region (all sizes 16B-aligned multiples).
  float* wsf = (float*)d_ws;
  size_t off = 0;
  float* u_f  = wsf + off; off += (size_t)MT * E_DIM;   // silu(u)
  float* sg   = wsf + off; off += (size_t)MT * E_DIM;   // silu(gate)
  float* dl   = wsf + off; off += (size_t)MT * E_DIM;   // delta
  float* Bm   = wsf + off; off += (size_t)MT * N_ST;
  float* Cm   = wsf + off; off += (size_t)MT * N_ST;

  __bf16* wsb = (__bf16*)(wsf + off);
  size_t ob = 0;
  __bf16* xn_b  = wsb + ob; ob += (size_t)MT * H_DIM;
  __bf16* u_b   = wsb + ob; ob += (size_t)MT * E_DIM;
  __bf16* dr_b  = wsb + ob; ob += (size_t)MT * R_DIM;
  __bf16* y_b   = wsb + ob; ob += (size_t)MT * E_DIM;
  __bf16* WinT  = wsb + ob; ob += (size_t)(2 * E_DIM) * H_DIM;  // [4096,1024]
  __bf16* WdelT = wsb + ob; ob += (size_t)R_DIM * E_DIM;        // [64,2048]
  __bf16* WdtT  = wsb + ob; ob += (size_t)E_DIM * R_DIM;        // [2048,64]
  __bf16* WBT   = wsb + ob; ob += (size_t)BN * E_DIM;           // [64,2048] pad
  __bf16* WCT   = wsb + ob; ob += (size_t)BN * E_DIM;           // [64,2048] pad
  __bf16* WoutT = wsb + ob; ob += (size_t)H_DIM * E_DIM;        // [1024,2048]

  const dim3 blk(256);

  // ---- weight prep: bf16 + transpose (+ zero-pad N to 64) -----------------
  cast_transpose<<<dim3(H_DIM/32, (2*E_DIM)/32), blk, 0, stream>>>(W_in,    WinT,  H_DIM, 2*E_DIM);
  cast_transpose<<<dim3(E_DIM/32, R_DIM/32),     blk, 0, stream>>>(W_delta, WdelT, E_DIM, R_DIM);
  cast_transpose<<<dim3(R_DIM/32, E_DIM/32),     blk, 0, stream>>>(W_dt,    WdtT,  R_DIM, E_DIM);
  cast_transpose<<<dim3(E_DIM/32, BN/32),        blk, 0, stream>>>(W_B,     WBT,   E_DIM, N_ST);
  cast_transpose<<<dim3(E_DIM/32, BN/32),        blk, 0, stream>>>(W_C,     WCT,   E_DIM, N_ST);
  cast_transpose<<<dim3(E_DIM/32, H_DIM/32),     blk, 0, stream>>>(W_out,   WoutT, E_DIM, H_DIM);

  // 1) LayerNorm -> bf16
  ln_kernel<<<dim3(MT), blk, 0, stream>>>(x, ln_gamma, ln_beta, xn_b);

  // 2) x_proj = xn @ W_in + b_in -> silu(u) [f32+bf16], silu(gate) [f32]
  wmma_gemm<1><<<dim3(MT/BM, (2*E_DIM)/BN), blk, 0, stream>>>(
      xn_b, WinT, b_in, u_f, sg, u_b, nullptr, MT, 2*E_DIM, H_DIM);

  // 3) dr = u @ W_delta + b_delta  [MT,64] bf16
  wmma_gemm<4><<<dim3(MT/BM, R_DIM/BN), blk, 0, stream>>>(
      u_b, WdelT, b_delta, nullptr, nullptr, dr_b, nullptr, MT, R_DIM, E_DIM);

  // 4) delta = softplus(dr @ W_dt + b_dt)  [MT,2048] f32
  wmma_gemm<2><<<dim3(MT/BM, E_DIM/BN), blk, 0, stream>>>(
      dr_b, WdtT, b_dt, dl, nullptr, nullptr, nullptr, MT, E_DIM, R_DIM);

  // 5) Bm / Cm  [MT,16] f32
  wmma_gemm<0><<<dim3(MT/BM, 1), blk, 0, stream>>>(
      u_b, WBT, b_B, Bm, nullptr, nullptr, nullptr, MT, N_ST, E_DIM);
  wmma_gemm<0><<<dim3(MT/BM, 1), blk, 0, stream>>>(
      u_b, WCT, b_C, Cm, nullptr, nullptr, nullptr, MT, N_ST, E_DIM);

  // 6) selective scan + D skip + gate -> y (bf16)
  scan_kernel<<<dim3((B_SZ*E_DIM)/256), blk, 0, stream>>>(
      u_f, dl, Bm, Cm, A_log, Dvec, sg, y_b);

  // 7) out = y @ W_out + b_out + residual
  wmma_gemm<3><<<dim3(MT/BM, H_DIM/BN), blk, 0, stream>>>(
      y_b, WoutT, b_out, out, nullptr, nullptr, x, MT, H_DIM, E_DIM);
}
